// FusedMistralLayerIntraInterFullFuse_80908593922053
// MI455X (gfx1250) — matrix operand order repaired and checked by reading, not compiled, not run on hardware
//
#include <hip/hip_runtime.h>
#include <cstdint>
#include <cstddef>

// ---------------------------------------------------------------------------
// CDNA5 / gfx1250 fused Mistral layer (RMSNorm + LoRA QKV + RoPE + flash GQA
// attention + LoRA O + RMSNorm + LoRA MLP), bf16 WMMA compute, fp32 I/O.
// ---------------------------------------------------------------------------

typedef __bf16 bf16_t;
typedef __attribute__((ext_vector_type(16))) __bf16 v16bf;
typedef __attribute__((ext_vector_type(8)))  float  v8f;

#define WMMA_BF16(a, b, c) \
  __builtin_amdgcn_wmma_f32_16x16x32_bf16(false, (a), false, (b), (short)0, (c), false, false)

namespace cfg {
constexpr int B = 2, S = 2048, D = 2048, H = 16, HKV = 8, HD = 128;
constexpr int DKV = 1024, DFF = 8192, R = 16;
constexpr int M = B * S;            // 4096 token rows
constexpr float EPS = 1e-5f;
constexpr float NEG = -1e9f;
}

// ---------------------------------------------------------------------------
// Fragment helpers (A/B use contiguous K order in both, C/D per ISA, wave32)
// ---------------------------------------------------------------------------

// A-matrix 16x32 bf16 fragment. Row = lane%16. K contiguous:
// vgpr v, half = lane/16: K = half*16 + v*2 (+0/+1)
// (16 contiguous bf16 per lane -> compiler merges into b128 loads)
__device__ inline v16bf load_a_frag(const bf16_t* A, int lda, int m0, int k0) {
  const int lane = threadIdx.x & 31;
  const int hl = lane >> 4;
  const bf16_t* p = A + (size_t)(m0 + (lane & 15)) * lda + k0;
  union { v16bf v; unsigned u[8]; } r;
#pragma unroll
  for (int v = 0; v < 8; ++v) {
    int k = hl * 16 + v * 2;
    r.u[v] = *(const unsigned*)(p + k);
  }
  return r.v;
}

// B-matrix 32x16 bf16 fragment, source stored column-major-for-B (i.e. Wt[N][K]).
// Col = lane%16; lanes 0-15 hold K=k0..k0+15, lanes 16-31 hold K=k0+16..k0+31,
// element j <-> K = half*16 + j (contiguous 32 bytes per lane).
__device__ inline v16bf load_b_frag(const bf16_t* Wt, int ldb, int n0, int k0) {
  const int lane = threadIdx.x & 31;
  const int hl = lane >> 4;
  const bf16_t* p = Wt + (size_t)(n0 + (lane & 15)) * ldb + k0 + hl * 16;
  union { v16bf v; uint4 q[2]; } r;
  r.q[0] = *(const uint4*)(p);
  r.q[1] = *(const uint4*)(p + 8);
  return r.v;
}

// C/D 16x16 f32: row = r + 8*(lane/16), col = lane%16
__device__ inline void store_c_frag(float* C, int ldc, int m0, int n0, v8f c) {
  const int lane = threadIdx.x & 31;
  const int hl = lane >> 4;
  const int cl = lane & 15;
#pragma unroll
  for (int r = 0; r < 8; ++r)
    C[(size_t)(m0 + r + 8 * hl) * ldc + n0 + cl] = c[r];
}

__device__ inline v8f zero8() {
  v8f z = {0.f, 0.f, 0.f, 0.f, 0.f, 0.f, 0.f, 0.f};
  return z;
}

// ---------------------------------------------------------------------------
// Utility kernels
// ---------------------------------------------------------------------------

// fp32 [K][N] -> bf16 transposed [N][K], 32x32 LDS tile, coalesced both sides.
__global__ __launch_bounds__(256)
void k_convT(const float* __restrict__ w, bf16_t* __restrict__ wt,
             int Kd, int Nd) {
  __shared__ float tile[32][33];          // +1 pad: no bank conflicts
  const int n0 = blockIdx.x * 32;
  const int k0 = blockIdx.y * 32;
  const int tx = threadIdx.x & 31;
  const int ty = threadIdx.x >> 5;        // 0..7
#pragma unroll
  for (int i = 0; i < 4; ++i)
    tile[ty + 8 * i][tx] = w[(size_t)(k0 + ty + 8 * i) * Nd + n0 + tx];
  __syncthreads();
#pragma unroll
  for (int i = 0; i < 4; ++i)
    wt[(size_t)(n0 + ty + 8 * i) * Kd + k0 + tx] = (bf16_t)tile[tx][ty + 8 * i];
}

__global__ void k_convert(const float* __restrict__ in, bf16_t* __restrict__ out,
                          size_t n) {
  size_t idx = (size_t)blockIdx.x * 256 + threadIdx.x;
  if (idx < n) out[idx] = (bf16_t)in[idx];
}

// RMSNorm: one block (256 threads) per row; fp32 in -> bf16 out
__global__ void k_rmsnorm(const float* __restrict__ x, const float* __restrict__ w,
                          bf16_t* __restrict__ out, int Dd) {
  const int row = blockIdx.x;
  const float* xr = x + (size_t)row * Dd;
  __shared__ float red[256];
  float ss = 0.f;
  for (int i = threadIdx.x; i < Dd; i += 256) { float v = xr[i]; ss += v * v; }
  red[threadIdx.x] = ss;
  __syncthreads();
  for (int st = 128; st > 0; st >>= 1) {
    if ((int)threadIdx.x < st) red[threadIdx.x] += red[threadIdx.x + st];
    __syncthreads();
  }
  float inv = rsqrtf(red[0] / (float)Dd + cfg::EPS);
  for (int i = threadIdx.x; i < Dd; i += 256)
    out[(size_t)row * Dd + i] = (bf16_t)(xr[i] * inv * w[i]);
}

// t[M][16] = X_bf16[M][K] @ a[K][16].  One block per token row:
// 16 K-slices x 16 ranks, LDS reduce over slices, serial loop in 16 threads.
__global__ __launch_bounds__(256)
void k_lora_xa(const bf16_t* __restrict__ X, const float* __restrict__ a,
               float* __restrict__ t, int Kd) {
  const int m = blockIdx.x;
  const int r = threadIdx.x & 15;
  const int ch = threadIdx.x >> 4;        // 0..15
  const int slice = Kd >> 4;              // K/16 (128 or 512)
  const bf16_t* xr = X + (size_t)m * Kd + (size_t)ch * slice;
  const float* ar = a + (size_t)ch * slice * 16 + r;
  float acc = 0.f;
  for (int k = 0; k < slice; ++k) acc += (float)xr[k] * ar[(size_t)k * 16];
  __shared__ float red[256];
  red[ch * 16 + r] = acc;
  __syncthreads();
  if (threadIdx.x < 16) {
    float s = 0.f;
#pragma unroll
    for (int c = 0; c < 16; ++c) s += red[c * 16 + threadIdx.x];
    t[m * 16 + threadIdx.x] = s;
  }
}

// out = C + t@bW (+ res). out may alias C.
__global__ void k_epilogue(const float* __restrict__ C, const float* __restrict__ t,
                           const float* __restrict__ bW, const float* __restrict__ res,
                           float* __restrict__ out, int Md, int Nd) {
  size_t total = (size_t)Md * Nd;
  size_t idx = (size_t)blockIdx.x * 256 + threadIdx.x;
  if (idx >= total) return;
  int n = (int)(idx % Nd);
  int m = (int)(idx / Nd);
  float acc = C[idx];
#pragma unroll
  for (int r = 0; r < 16; ++r) acc += t[m * 16 + r] * bW[(size_t)r * Nd + n];
  if (res) acc += res[idx];
  out[idx] = acc;
}

// RoPE + split into heads: fp32 [B,S,nh*HD] -> bf16 [B,nh,S,HD]
__global__ void k_rope_heads(const float* __restrict__ in,
                             const float* __restrict__ cosb,
                             const float* __restrict__ sinb,
                             bf16_t* __restrict__ out, int nh) {
  using namespace cfg;
  size_t idx = (size_t)blockIdx.x * 256 + threadIdx.x;
  size_t total = (size_t)B * nh * S * (HD / 2);
  if (idx >= total) return;
  int i = (int)(idx & 63);            // HD/2 = 64
  size_t rest = idx >> 6;
  int s = (int)(rest % S); rest /= S;
  int hd = (int)(rest % nh);
  int b = (int)(rest / nh);
  const float* row = in + ((size_t)b * S + s) * (nh * HD) + (size_t)hd * HD;
  float x1 = row[i];
  float x2 = row[i + 64];
  float c = cosb[(size_t)s * HD + i];
  float sn = sinb[(size_t)s * HD + i];
  bf16_t* orow = out + (((size_t)b * nh + hd) * S + s) * HD;
  orow[i]      = (bf16_t)(x1 * c - x2 * sn);
  orow[i + 64] = (bf16_t)(x2 * c + x1 * sn);
}

// V: fp32 [B,S,HKV*HD] -> bf16 transposed per head [B,HKV,HD,S]
__global__ void k_v_transpose(const float* __restrict__ in, bf16_t* __restrict__ out) {
  using namespace cfg;
  size_t idx = (size_t)blockIdx.x * 256 + threadIdx.x;
  size_t total = (size_t)B * HKV * S * HD;
  if (idx >= total) return;
  int hd = (int)(idx % HD);
  size_t rest = idx / HD;
  int s = (int)(rest % S); rest /= S;
  int h = (int)(rest % HKV);
  int b = (int)(rest / HKV);
  out[(((size_t)b * HKV + h) * HD + hd) * S + s] =
      (bf16_t)in[((size_t)b * S + s) * DKV + (size_t)h * HD + hd];
}

__global__ void k_silu_mul(const float* __restrict__ g, const float* __restrict__ u,
                           bf16_t* __restrict__ out, size_t n) {
  size_t idx = (size_t)blockIdx.x * 256 + threadIdx.x;
  if (idx >= n) return;
  float gv = g[idx];
  float sg = gv / (1.f + __expf(-gv));
  out[idx] = (bf16_t)(sg * u[idx]);
}

// ---------------------------------------------------------------------------
// WMMA GEMM: C[M][N] (fp32) = A_bf16[M][K] @ Wt_bf16[N][K]^T
// Block = 8 waves (256 thr), wave tile 32x64 (2x4 WMMA, 8 wmma / 6 frag loads),
// block tile 64x256.  global_prefetch two k-steps ahead.
// ---------------------------------------------------------------------------
__global__ __launch_bounds__(256)
void k_gemm(const bf16_t* __restrict__ A, const bf16_t* __restrict__ Wt,
            float* __restrict__ C, int Md, int Nd, int Kd) {
  const int wave = threadIdx.x >> 5;
  const int wm = wave >> 2;          // 0..1
  const int wn = wave & 3;           // 0..3
  const int m0 = blockIdx.x * 64 + wm * 32;
  const int n0 = blockIdx.y * 256 + wn * 64;
  const int lane = threadIdx.x & 31;

  v8f c00 = zero8(), c01 = zero8(), c02 = zero8(), c03 = zero8();
  v8f c10 = zero8(), c11 = zero8(), c12 = zero8(), c13 = zero8();

  for (int k0 = 0; k0 < Kd; k0 += 32) {
    if (k0 + 64 < Kd) {   // prefetch 2 steps ahead (global_prefetch_b8)
      __builtin_prefetch(A + (size_t)(m0 + lane) * Kd + k0 + 64, 0, 1);
      __builtin_prefetch(Wt + (size_t)(n0 + lane) * Kd + k0 + 64, 0, 1);
      __builtin_prefetch(Wt + (size_t)(n0 + 32 + lane) * Kd + k0 + 64, 0, 1);
    }
    v16bf a0 = load_a_frag(A, Kd, m0, k0);
    v16bf a1 = load_a_frag(A, Kd, m0 + 16, k0);
    v16bf b0 = load_b_frag(Wt, Kd, n0, k0);
    v16bf b1 = load_b_frag(Wt, Kd, n0 + 16, k0);
    v16bf b2 = load_b_frag(Wt, Kd, n0 + 32, k0);
    v16bf b3 = load_b_frag(Wt, Kd, n0 + 48, k0);
    c00 = WMMA_BF16(a0, b0, c00);
    c01 = WMMA_BF16(a0, b1, c01);
    c02 = WMMA_BF16(a0, b2, c02);
    c03 = WMMA_BF16(a0, b3, c03);
    c10 = WMMA_BF16(a1, b0, c10);
    c11 = WMMA_BF16(a1, b1, c11);
    c12 = WMMA_BF16(a1, b2, c12);
    c13 = WMMA_BF16(a1, b3, c13);
  }
  store_c_frag(C, Nd, m0, n0,      c00);
  store_c_frag(C, Nd, m0, n0 + 16, c01);
  store_c_frag(C, Nd, m0, n0 + 32, c02);
  store_c_frag(C, Nd, m0, n0 + 48, c03);
  store_c_frag(C, Nd, m0 + 16, n0,      c10);
  store_c_frag(C, Nd, m0 + 16, n0 + 16, c11);
  store_c_frag(C, Nd, m0 + 16, n0 + 32, c12);
  store_c_frag(C, Nd, m0 + 16, n0 + 48, c13);
}

// ---------------------------------------------------------------------------
// Flash attention (causal, GQA). One wave per (b, h, 16-query tile).
// Q:[B,H,S,HD] bf16, K:[B,HKV,S,HD] bf16, Vt:[B,HKV,HD,S] bf16, O:[B,S,H,HD] f32
// ---------------------------------------------------------------------------
__global__ __launch_bounds__(32)
void k_attention(const bf16_t* __restrict__ Q, const bf16_t* __restrict__ Km,
                 const bf16_t* __restrict__ Vt, float* __restrict__ O) {
  using namespace cfg;
  const int lane = threadIdx.x & 31;
  const int hl = lane >> 4;
  const int cl = lane & 15;
  int bid = blockIdx.x;
  const int QT = S / 16;
  int qt = bid % QT; bid /= QT;
  int h = bid % H;
  int b = bid / H;
  int kvh = h / (H / HKV);
  int q0 = qt * 16;

  const bf16_t* qb = Q + (((size_t)b * H + h) * S) * HD;
  const bf16_t* kb = Km + (((size_t)b * HKV + kvh) * S) * HD;
  const bf16_t* vb = Vt + (((size_t)b * HKV + kvh) * HD) * S;

  v16bf qa[4];
#pragma unroll
  for (int d = 0; d < 4; ++d) qa[d] = load_a_frag(qb, HD, q0, d * 32);

  v8f oacc[8];
#pragma unroll
  for (int f = 0; f < 8; ++f) oacc[f] = zero8();
  float m_i[8], l_i[8];
#pragma unroll
  for (int r = 0; r < 8; ++r) { m_i[r] = -3.0e38f; l_i[r] = 0.f; }

  __shared__ __align__(16) bf16_t p_lds[16 * 32];
  const float scale = 0.08838834764831845f;   // 1/sqrt(128)

  const int kb_end = (q0 + 15) / 32;
  for (int kblk = 0; kblk <= kb_end; ++kblk) {
    const int kk0 = kblk * 32;
    // ---- scores S(16x32) = Q(16x128) @ K^T ----
    v8f s0 = zero8(), s1 = zero8();
#pragma unroll
    for (int d = 0; d < 4; ++d) {
      v16bf bk = load_b_frag(kb, HD, kk0, d * 32);
      s0 = WMMA_BF16(qa[d], bk, s0);
    }
#pragma unroll
    for (int d = 0; d < 4; ++d) {
      v16bf bk = load_b_frag(kb, HD, kk0 + 16, d * 32);
      s1 = WMMA_BF16(qa[d], bk, s1);
    }
    // ---- online softmax (rows split across 16-lane halves) ----
#pragma unroll
    for (int r = 0; r < 8; ++r) {
      int qrow = q0 + r + 8 * hl;
      float a0 = (kk0 + cl <= qrow)      ? s0[r] * scale : NEG;
      float a1 = (kk0 + 16 + cl <= qrow) ? s1[r] * scale : NEG;
      float lm = fmaxf(a0, a1);
      lm = fmaxf(lm, __shfl_xor(lm, 1));
      lm = fmaxf(lm, __shfl_xor(lm, 2));
      lm = fmaxf(lm, __shfl_xor(lm, 4));
      lm = fmaxf(lm, __shfl_xor(lm, 8));
      float mnew = fmaxf(m_i[r], lm);
      float corr = __expf(m_i[r] - mnew);
      float p0 = __expf(a0 - mnew);
      float p1 = __expf(a1 - mnew);
      float ls = p0 + p1;
      ls += __shfl_xor(ls, 1);
      ls += __shfl_xor(ls, 2);
      ls += __shfl_xor(ls, 4);
      ls += __shfl_xor(ls, 8);
      l_i[r] = l_i[r] * corr + ls;
      m_i[r] = mnew;
#pragma unroll
      for (int f = 0; f < 8; ++f) oacc[f][r] *= corr;
      p_lds[(r + 8 * hl) * 32 + cl]      = (bf16_t)p0;
      p_lds[(r + 8 * hl) * 32 + 16 + cl] = (bf16_t)p1;
    }
    __syncthreads();
    // ---- O(16x128) += P(16x32) @ V(32x128) ----
    v16bf pa = load_a_frag(p_lds, 32, 0, 0);
#pragma unroll
    for (int f = 0; f < 8; ++f) {
      v16bf bv = load_b_frag(vb, S, f * 16, kk0);
      oacc[f] = WMMA_BF16(pa, bv, oacc[f]);
    }
    __syncthreads();
  }
  // ---- normalize + store to [B,S,H,HD] ----
#pragma unroll
  for (int r = 0; r < 8; ++r) {
    float inv = 1.0f / l_i[r];
    int srow = q0 + r + 8 * hl;
    float* orow = O + ((size_t)(b * S + srow) * H + h) * HD;
#pragma unroll
    for (int f = 0; f < 8; ++f) orow[f * 16 + cl] = oacc[f][r] * inv;
  }
}

// ---------------------------------------------------------------------------
// Launch
// ---------------------------------------------------------------------------
extern "C" void kernel_launch(void* const* d_in, const int* in_sizes, int n_in,
                              void* d_out, int out_size, void* d_ws, size_t ws_size,
                              hipStream_t stream) {
  using namespace cfg;
  const float* x    = (const float*)d_in[0];
  const float* nw1  = (const float*)d_in[1];
  const float* nw2  = (const float*)d_in[2];
  const float* cosb = (const float*)d_in[3];
  const float* sinb = (const float*)d_in[4];
  // d_in[5] attention_mask: pure causal, computed analytically in-kernel.
  const float* w_q  = (const float*)d_in[6];
  const float* a_q  = (const float*)d_in[7];
  const float* b_q  = (const float*)d_in[8];
  const float* w_k  = (const float*)d_in[9];
  const float* a_k  = (const float*)d_in[10];
  const float* b_k  = (const float*)d_in[11];
  const float* w_v  = (const float*)d_in[12];
  const float* a_v  = (const float*)d_in[13];
  const float* b_v  = (const float*)d_in[14];
  const float* w_o  = (const float*)d_in[15];
  const float* a_o  = (const float*)d_in[16];
  const float* b_o  = (const float*)d_in[17];
  const float* w_g  = (const float*)d_in[18];
  const float* a_g  = (const float*)d_in[19];
  const float* b_g  = (const float*)d_in[20];
  const float* w_u  = (const float*)d_in[21];
  const float* a_u  = (const float*)d_in[22];
  const float* b_u  = (const float*)d_in[23];
  const float* w_d  = (const float*)d_in[24];
  const float* a_d  = (const float*)d_in[25];
  const float* b_d  = (const float*)d_in[26];
  float* out = (float*)d_out;

  // ---- workspace carve (phase-reused) ----
  char* ws = (char*)d_ws;
  size_t off = 0;
  auto alloc = [&](size_t bytes) -> char* {
    char* p = ws + off;
    off = (off + bytes + 255) & ~(size_t)255;
    return p;
  };
  bf16_t* wq_t = (bf16_t*)alloc((size_t)D * D * 2);
  bf16_t* wk_t = (bf16_t*)alloc((size_t)D * DKV * 2);
  bf16_t* wv_t = (bf16_t*)alloc((size_t)D * DKV * 2);
  bf16_t* wo_t = (bf16_t*)alloc((size_t)D * D * 2);
  bf16_t* wg_t = (bf16_t*)alloc((size_t)D * DFF * 2);
  bf16_t* wu_t = (bf16_t*)alloc((size_t)D * DFF * 2);
  bf16_t* wd_t = (bf16_t*)alloc((size_t)DFF * D * 2);
  bf16_t* xn_bf = (bf16_t*)alloc((size_t)M * D * 2);   // xn1 then xn2
  float*  t_lr  = (float*)alloc((size_t)M * R * 4);
  char* bufA = alloc((size_t)M * DFF * 4);  // qkv f32 -> oC -> gate -> downC
  char* bufB = alloc((size_t)M * DFF * 4);  // q/k/v bf16 heads -> up
  char* bufC = alloc((size_t)M * DFF * 2);  // O f32 + o bf16 -> h bf16
  float* x_med = (float*)alloc((size_t)M * D * 4);

  float* qf = (float*)bufA;
  float* kf = (float*)(bufA + (size_t)M * D * 4);
  float* vf = (float*)(bufA + (size_t)M * D * 4 + (size_t)M * DKV * 4);
  bf16_t* qbf = (bf16_t*)bufB;
  bf16_t* kbf = (bf16_t*)(bufB + (size_t)M * D * 2);
  bf16_t* vtb = (bf16_t*)(bufB + (size_t)M * D * 2 + (size_t)M * DKV * 2);
  float* Obuf = (float*)bufC;
  bf16_t* obf = (bf16_t*)(bufC + (size_t)M * D * 4);
  float* oC    = (float*)bufA;     // q/k/v f32 dead after RoPE
  float* gateC = (float*)bufA;     // oC dead after x_med epilogue
  float* upC   = (float*)bufB;     // head buffers dead after attention
  bf16_t* hbf  = (bf16_t*)bufC;    // O/o dead after o-projection
  float* dC    = (float*)bufA;     // gate dead after silu

  dim3 blk(256);
  auto g1 = [](size_t total) { return dim3((unsigned)((total + 255) / 256)); };

  // 1) weight converts (fp32 [K][N] -> bf16 [N][K]), tiled transpose
  k_convT<<<dim3(D / 32, D / 32), blk, 0, stream>>>(w_q, wq_t, D, D);
  k_convT<<<dim3(DKV / 32, D / 32), blk, 0, stream>>>(w_k, wk_t, D, DKV);
  k_convT<<<dim3(DKV / 32, D / 32), blk, 0, stream>>>(w_v, wv_t, D, DKV);
  k_convT<<<dim3(D / 32, D / 32), blk, 0, stream>>>(w_o, wo_t, D, D);
  k_convT<<<dim3(DFF / 32, D / 32), blk, 0, stream>>>(w_g, wg_t, D, DFF);
  k_convT<<<dim3(DFF / 32, D / 32), blk, 0, stream>>>(w_u, wu_t, D, DFF);
  k_convT<<<dim3(D / 32, DFF / 32), blk, 0, stream>>>(w_d, wd_t, DFF, D);

  // 2) xn1 = rmsnorm(x)
  k_rmsnorm<<<M, blk, 0, stream>>>(x, nw1, xn_bf, D);

  // 3) Q/K/V projections (+LoRA)
  k_gemm<<<dim3(M / 64, D / 256), blk, 0, stream>>>(xn_bf, wq_t, qf, M, D, D);
  k_lora_xa<<<M, blk, 0, stream>>>(xn_bf, a_q, t_lr, D);
  k_epilogue<<<g1((size_t)M * D), blk, 0, stream>>>(qf, t_lr, b_q, nullptr, qf, M, D);

  k_gemm<<<dim3(M / 64, DKV / 256), blk, 0, stream>>>(xn_bf, wk_t, kf, M, DKV, D);
  k_lora_xa<<<M, blk, 0, stream>>>(xn_bf, a_k, t_lr, D);
  k_epilogue<<<g1((size_t)M * DKV), blk, 0, stream>>>(kf, t_lr, b_k, nullptr, kf, M, DKV);

  k_gemm<<<dim3(M / 64, DKV / 256), blk, 0, stream>>>(xn_bf, wv_t, vf, M, DKV, D);
  k_lora_xa<<<M, blk, 0, stream>>>(xn_bf, a_v, t_lr, D);
  k_epilogue<<<g1((size_t)M * DKV), blk, 0, stream>>>(vf, t_lr, b_v, nullptr, vf, M, DKV);

  // 4) RoPE + head layout (V transposed per head)
  k_rope_heads<<<g1((size_t)B * H * S * (HD / 2)), blk, 0, stream>>>(qf, cosb, sinb, qbf, H);
  k_rope_heads<<<g1((size_t)B * HKV * S * (HD / 2)), blk, 0, stream>>>(kf, cosb, sinb, kbf, HKV);
  k_v_transpose<<<g1((size_t)B * HKV * S * HD), blk, 0, stream>>>(vf, vtb);

  // 5) flash attention
  k_attention<<<B * H * (S / 16), 32, 0, stream>>>(qbf, kbf, vtb, Obuf);

  // 6) O projection (+LoRA) + residual -> x_medium
  k_convert<<<g1((size_t)M * D), blk, 0, stream>>>(Obuf, obf, (size_t)M * D);
  k_gemm<<<dim3(M / 64, D / 256), blk, 0, stream>>>(obf, wo_t, oC, M, D, D);
  k_lora_xa<<<M, blk, 0, stream>>>(obf, a_o, t_lr, D);
  k_epilogue<<<g1((size_t)M * D), blk, 0, stream>>>(oC, t_lr, b_o, x, x_med, M, D);

  // 7) xn2 = rmsnorm(x_medium)
  k_rmsnorm<<<M, blk, 0, stream>>>(x_med, nw2, xn_bf, D);

  // 8) MLP gate/up (+LoRA)
  k_gemm<<<dim3(M / 64, DFF / 256), blk, 0, stream>>>(xn_bf, wg_t, gateC, M, DFF, D);
  k_lora_xa<<<M, blk, 0, stream>>>(xn_bf, a_g, t_lr, D);
  k_epilogue<<<g1((size_t)M * DFF), blk, 0, stream>>>(gateC, t_lr, b_g, nullptr, gateC, M, DFF);

  k_gemm<<<dim3(M / 64, DFF / 256), blk, 0, stream>>>(xn_bf, wu_t, upC, M, DFF, D);
  k_lora_xa<<<M, blk, 0, stream>>>(xn_bf, a_u, t_lr, D);
  k_epilogue<<<g1((size_t)M * DFF), blk, 0, stream>>>(upC, t_lr, b_u, nullptr, upC, M, DFF);

  // 9) h = silu(gate)*up, then down (+LoRA) + residual -> out
  k_silu_mul<<<g1((size_t)M * DFF), blk, 0, stream>>>(gateC, upC, hbf, (size_t)M * DFF);
  k_gemm<<<dim3(M / 64, D / 256), blk, 0, stream>>>(hbf, wd_t, dC, M, D, DFF);
  k_lora_xa<<<M, blk, 0, stream>>>(hbf, a_d, t_lr, DFF);
  k_epilogue<<<g1((size_t)M * D), blk, 0, stream>>>(dC, t_lr, b_d, x_med, out, M, D);

  (void)in_sizes; (void)n_in; (void)out_size; (void)ws_size;
}
